// AttentiveFPModel_20856361189761
// MI455X (gfx1250) — matrix operand order repaired and checked by reading, not compiled, not run on hardware
//
#include <hip/hip_runtime.h>
#include <hip/hip_fp16.h>

typedef __attribute__((ext_vector_type(16))) _Float16 v16h;
typedef __attribute__((ext_vector_type(8)))  _Float16 v8h;
typedef __attribute__((ext_vector_type(8)))  float    v8f;

__device__ __forceinline__ float lrelu(float x){ return x > 0.f ? x : 0.01f * x; }
__device__ __forceinline__ float eluf (float x){ return x > 0.f ? x : (expf(x) - 1.f); }
__device__ __forceinline__ float sigm (float x){ return 1.f / (1.f + expf(-x)); }

// ---------------- WMMA fragment helpers (ISA 7.12.2 layouts, wave32) ----------------
// B fragment for v_wmma_f32_16x16x32_f16. W row-major [NC, ldk] f16, B[k][n] = W[n][k].
// Lane: col = lane&15, half = lane>>4 ; b[0..7] = W[n0+col][kb + half*8 + i], b[8..15] = W[n0+col][kb + 16 + half*8 + i].
__device__ __forceinline__ v16h load_bfrag(const _Float16* __restrict__ W, int ldk,
                                           int n0, int kb, int lane) {
    const _Float16* p = W + (size_t)(n0 + (lane & 15)) * ldk + kb + ((lane >> 4) << 3);
    v8h lo = *(const v8h*)p;
    v8h hi = *(const v8h*)(p + 16);
    v16h b;
#pragma unroll
    for (int i = 0; i < 8; ++i) { b[i] = lo[i]; b[8 + i] = hi[i]; }
    return b;
}

// A fragment from f32 row-major [M, ldk]. Lane: row = m0+(lane&15), half = lane>>4.
// a[0..7] = A[row][kb + half*8 + i], a[8..15] = A[row][kb + 16 + half*8 + i].
template<int ACT>  // 0 = none, 1 = elu
__device__ __forceinline__ v16h load_afrag_f32(const float* __restrict__ A, int ldk,
                                               int m0, int kb, int lane) {
    const float* p = A + (size_t)(m0 + (lane & 15)) * ldk + kb + ((lane >> 4) << 3);
    v8f lo = *(const v8f*)p;
    v8f hi = *(const v8f*)(p + 16);
    v16h a;
#pragma unroll
    for (int i = 0; i < 8; ++i) {
        float x = lo[i], y = hi[i];
        if (ACT == 1) { x = eluf(x); y = eluf(y); }
        a[i] = (_Float16)x; a[8 + i] = (_Float16)y;
    }
    return a;
}

__device__ __forceinline__ v16h load_afrag_f16(const _Float16* __restrict__ A, int ldk,
                                               int m0, int kb, int lane) {
    const _Float16* p = A + (size_t)(m0 + (lane & 15)) * ldk + kb + ((lane >> 4) << 3);
    v8h lo = *(const v8h*)p;
    v8h hi = *(const v8h*)(p + 16);
    v16h a;
#pragma unroll
    for (int i = 0; i < 8; ++i) { a[i] = lo[i]; a[8 + i] = hi[i]; }
    return a;
}

// ---------------- Generic GEMM: C[M,NC] = act(A[M,K] @ W[NC,K]^T + bias) ----------------
template<int K, int NC, int ACT_A, int ACT_OUT>
__global__ void gemm_nt_kernel(const float* __restrict__ A, const _Float16* __restrict__ W,
                               const float* __restrict__ bias, float* __restrict__ C, int M) {
    const int lane = threadIdx.x & 31;
    const int tile = blockIdx.x * (blockDim.x >> 5) + (threadIdx.x >> 5);
    const int m0 = tile << 4;
    if (m0 >= M) return;
    constexpr int NT = NC / 16;
    v8f acc[NT] = {};
#pragma unroll
    for (int kb = 0; kb < K; kb += 32) {
        if (kb + 32 < K)
            __builtin_prefetch(A + (size_t)m0 * K + kb + 32, 0, 1);  // global_prefetch_b8
        v16h a = load_afrag_f32<ACT_A>(A, K, m0, kb, lane);
#pragma unroll
        for (int t = 0; t < NT; ++t) {
            v16h b = load_bfrag(W, K, t * 16, kb, lane);
            acc[t] = __builtin_amdgcn_wmma_f32_16x16x32_f16(false, a, false, b,
                                                            (short)0, acc[t], false, false);
        }
    }
    const int col = lane & 15;
    const int rbase = m0 + ((lane >> 4) << 3);
#pragma unroll
    for (int t = 0; t < NT; ++t) {
        const int n = t * 16 + col;
        const float bv = bias ? bias[n] : 0.f;
#pragma unroll
        for (int r = 0; r < 8; ++r) {
            float v = acc[t][r] + bv;
            if (ACT_OUT == 1) v = lrelu(v);
            C[(size_t)(rbase + r) * NC + n] = v;
        }
    }
}

// ------------- GATEConv edge GEMM: m = lrelu(concat(x0[src], eattr) @ Wg1^T), K padded to 128
// also computes logit[e] = lrelu(m . g_att_l + xr[dst])
__global__ void gate_edge_kernel(const float* __restrict__ x0, const float* __restrict__ eattr,
                                 const int* __restrict__ srcI, const int* __restrict__ dstI,
                                 const _Float16* __restrict__ Wg1, const float* __restrict__ attl,
                                 const float* __restrict__ xr, _Float16* __restrict__ m_out,
                                 float* __restrict__ logit, int E) {
    const int lane = threadIdx.x & 31;
    const int tile = blockIdx.x * (blockDim.x >> 5) + (threadIdx.x >> 5);
    const int e0 = tile << 4;
    if (e0 >= E) return;
    const int half = lane >> 4;
    const int erow = e0 + (lane & 15);
    const int sn = srcI[erow];
    v8f acc[6] = {};
#pragma unroll
    for (int c = 0; c < 3; ++c) {                      // K chunks 0,32,64 from x0[src]
        const int kb = c * 32;
        const float* p = x0 + (size_t)sn * 96 + kb + (half << 3);
        v8f lo = *(const v8f*)p;
        v8f hi = *(const v8f*)(p + 16);
        v16h a;
#pragma unroll
        for (int i = 0; i < 8; ++i) { a[i] = (_Float16)lo[i]; a[8 + i] = (_Float16)hi[i]; }
#pragma unroll
        for (int t = 0; t < 6; ++t) {
            v16h b = load_bfrag(Wg1, 128, t * 16, kb, lane);
            acc[t] = __builtin_amdgcn_wmma_f32_16x16x32_f16(false, a, false, b,
                                                            (short)0, acc[t], false, false);
        }
    }
    {   // K chunk 96..127: 14 edge_attr values + zero pad
        const float* pe = eattr + (size_t)erow * 14;
        v16h a;
#pragma unroll
        for (int i = 0; i < 8; ++i) {
            const int idx = (half << 3) + i;           // K-96 in 0..15
            a[i] = (_Float16)((idx < 14) ? pe[idx] : 0.f);
            a[8 + i] = (_Float16)0.f;                  // K 112..127 = 0
        }
#pragma unroll
        for (int t = 0; t < 6; ++t) {
            v16h b = load_bfrag(Wg1, 128, t * 16, 96, lane);
            acc[t] = __builtin_amdgcn_wmma_f32_16x16x32_f16(false, a, false, b,
                                                            (short)0, acc[t], false, false);
        }
    }
    const int col = lane & 15;
    const int rb = half << 3;
    float al[6];
#pragma unroll
    for (int t = 0; t < 6; ++t) al[t] = attl[t * 16 + col];
    float s[8];
#pragma unroll
    for (int r = 0; r < 8; ++r) s[r] = 0.f;
#pragma unroll
    for (int t = 0; t < 6; ++t) {
#pragma unroll
        for (int r = 0; r < 8; ++r) {
            const float v = lrelu(acc[t][r]);
            m_out[(size_t)(e0 + rb + r) * 96 + t * 16 + col] = (_Float16)v;
            s[r] += v * al[t];
        }
    }
#pragma unroll
    for (int r = 0; r < 8; ++r) {
#pragma unroll
        for (int mk = 1; mk < 16; mk <<= 1) s[r] += __shfl_xor(s[r], mk, 32);
    }
    if (col == 0) {
#pragma unroll
        for (int r = 0; r < 8; ++r) {
            const int e = e0 + rb + r;
            logit[e] = lrelu(s[r] + xr[dstI[e]]);
        }
    }
}

// ------------- GATEConv aggregation: h[dst] += (m @ Wg2^T) * alpha  (fused GEMM + scatter)
__global__ void gate_agg_kernel(const _Float16* __restrict__ m, const _Float16* __restrict__ Wg2,
                                const float* __restrict__ ew, const int* __restrict__ dstI,
                                float* __restrict__ hacc, int E) {
    const int lane = threadIdx.x & 31;
    const int tile = blockIdx.x * (blockDim.x >> 5) + (threadIdx.x >> 5);
    const int e0 = tile << 4;
    if (e0 >= E) return;
    v8f acc[6] = {};
#pragma unroll
    for (int kb = 0; kb < 96; kb += 32) {
        v16h a = load_afrag_f16(m, 96, e0, kb, lane);
#pragma unroll
        for (int t = 0; t < 6; ++t) {
            v16h b = load_bfrag(Wg2, 96, t * 16, kb, lane);
            acc[t] = __builtin_amdgcn_wmma_f32_16x16x32_f16(false, a, false, b,
                                                            (short)0, acc[t], false, false);
        }
    }
    const int col = lane & 15;
    const int rb = e0 + ((lane >> 4) << 3);
#pragma unroll
    for (int r = 0; r < 8; ++r) {
        const int e = rb + r;
        const float w = ew[e];
        const int d = dstI[e];
#pragma unroll
        for (int t = 0; t < 6; ++t)
            atomicAdd(&hacc[(size_t)d * 96 + t * 16 + col], acc[t][r] * w);
    }
}

// ---------------- small elementwise / segment kernels ----------------
__global__ void cvt_w_kernel(const float* __restrict__ s, _Float16* __restrict__ d,
                             int rows, int scols, int dcols) {
    int t = blockIdx.x * blockDim.x + threadIdx.x;
    if (t >= rows * dcols) return;
    int r = t / dcols, c = t % dcols;
    d[t] = (_Float16)((c < scols) ? s[(size_t)r * scols + c] : 0.f);
}

__global__ void zero_kernel(float* __restrict__ p, long n) {
    long t = (long)blockIdx.x * blockDim.x + threadIdx.x;
    if (t < n) p[t] = 0.f;
}

__global__ void rowdot_kernel(const float* __restrict__ X, const float* __restrict__ v,
                              float* __restrict__ out, int M, int D) {
    int i = blockIdx.x * blockDim.x + threadIdx.x;
    if (i >= M) return;
    const float* r = X + (size_t)i * D;
    float s = 0.f;
    for (int c = 0; c < D; ++c) s += r[c] * v[c];
    out[i] = s;
}

__global__ void pair_logit_kernel(const float* __restrict__ a, const float* __restrict__ b,
                                  const int* __restrict__ ia, const int* __restrict__ ib,
                                  float* __restrict__ logit, int n) {
    int i = blockIdx.x * blockDim.x + threadIdx.x;
    if (i >= n) return;
    logit[i] = lrelu(a[ia ? ia[i] : i] + b[ib[i]]);
}

__global__ void seg_init_kernel(unsigned* __restrict__ smax, float* __restrict__ ssum, int S) {
    int i = blockIdx.x * blockDim.x + threadIdx.x;
    if (i < S) { smax[i] = 0u; ssum[i] = 0.f; }
}
__global__ void seg_max_kernel(const float* __restrict__ lg, const int* __restrict__ seg,
                               unsigned* __restrict__ smax, int n) {
    int i = blockIdx.x * blockDim.x + threadIdx.x;
    if (i >= n) return;
    unsigned bits = __float_as_uint(lg[i]);
    unsigned key = (bits & 0x80000000u) ? ~bits : (bits | 0x80000000u);  // order-preserving
    atomicMax(&smax[seg[i]], key);
}
__global__ void seg_exp_kernel(const float* __restrict__ lg, const int* __restrict__ seg,
                               const unsigned* __restrict__ smax, float* __restrict__ e,
                               float* __restrict__ ssum, int n) {
    int i = blockIdx.x * blockDim.x + threadIdx.x;
    if (i >= n) return;
    const int s = seg[i];
    const unsigned u = smax[s];
    float mx = 0.f;
    if (u != 0u) {
        unsigned bits = (u & 0x80000000u) ? (u ^ 0x80000000u) : ~u;
        mx = __uint_as_float(bits);
        if (!isfinite(mx)) mx = 0.f;
    }
    const float v = expf(lg[i] - mx);
    e[i] = v;
    atomicAdd(&ssum[s], v);
}
__global__ void seg_norm_kernel(float* __restrict__ e, const int* __restrict__ seg,
                                const float* __restrict__ ssum, int n) {
    int i = blockIdx.x * blockDim.x + threadIdx.x;
    if (i >= n) return;
    e[i] = e[i] / (ssum[seg[i]] + 1e-16f);
}

// acc[seg[i]][c] += X[rowidx?rowidx[i]:i][c] * (wgt?wgt[i]:1) over 96 cols
__global__ void scatter_rows_kernel(const float* __restrict__ X, const int* __restrict__ rowidx,
                                    const int* __restrict__ seg, const float* __restrict__ wgt,
                                    float* __restrict__ acc, long total) {
    long t = (long)blockIdx.x * blockDim.x + threadIdx.x;
    if (t >= total) return;
    int i = (int)(t / 96), c = (int)(t % 96);
    int r = rowidx ? rowidx[i] : i;
    float w = wgt ? wgt[i] : 1.f;
    atomicAdd(&acc[(size_t)seg[i] * 96 + c], X[(size_t)r * 96 + c] * w);
}

__global__ void add_bias_kernel(float* __restrict__ X, const float* __restrict__ b, long total) {
    long t = (long)blockIdx.x * blockDim.x + threadIdx.x;
    if (t >= total) return;
    X[t] += b[t % 96];
}

__global__ void relu_kernel(float* __restrict__ X, long n) {
    long t = (long)blockIdx.x * blockDim.x + threadIdx.x;
    if (t < n) X[t] = fmaxf(X[t], 0.f);
}

// out = relu(GRU(gi + bih, gh + bhh, hprev)) ; gi/gh are [M,288], H = 96
__global__ void gru_kernel(const float* __restrict__ gi, const float* __restrict__ gh,
                           const float* __restrict__ bih, const float* __restrict__ bhh,
                           const float* __restrict__ hprev, float* __restrict__ out, int M) {
    long t = (long)blockIdx.x * blockDim.x + threadIdx.x;
    if (t >= (long)M * 96) return;
    int row = (int)(t / 96), c = (int)(t % 96);
    const float* gir = gi + (size_t)row * 288;
    const float* ghr = gh + (size_t)row * 288;
    float ir = gir[c] + bih[c],        hr = ghr[c] + bhh[c];
    float iz = gir[96 + c] + bih[96 + c],  hz = ghr[96 + c] + bhh[96 + c];
    float in = gir[192 + c] + bih[192 + c], hn = ghr[192 + c] + bhh[192 + c];
    float r = sigm(ir + hr);
    float z = sigm(iz + hz);
    float nn = tanhf(in + r * hn);
    float v = (1.f - z) * nn + z * hprev[(size_t)row * 96 + c];
    out[(size_t)row * 96 + c] = fmaxf(v, 0.f);
}

__global__ void final_dot_kernel(const float* __restrict__ out_g, const float* __restrict__ w,
                                 const float* __restrict__ b, float* __restrict__ y, int G) {
    int g = blockIdx.x * blockDim.x + threadIdx.x;
    if (g >= G) return;
    const float* r = out_g + (size_t)g * 96;
    float s = 0.f;
    for (int c = 0; c < 96; ++c) s += r[c] * w[c];
    y[g] = s + b[0];
}

// ---------------- host orchestration ----------------
extern "C" void kernel_launch(void* const* d_in, const int* in_sizes, int n_in,
                              void* d_out, int out_size, void* d_ws, size_t ws_size,
                              hipStream_t stream) {
    (void)n_in; (void)ws_size;
    const float* x_in    = (const float*)d_in[0];
    const float* eattr   = (const float*)d_in[1];
    const int*   eidx    = (const int*)  d_in[2];
    const int*   batch   = (const int*)  d_in[3];
    const float* lin1_w  = (const float*)d_in[4];
    const float* lin1_b  = (const float*)d_in[5];
    const float* g_att_l = (const float*)d_in[6];
    const float* g_att_r = (const float*)d_in[7];
    const float* g_lin1w = (const float*)d_in[8];
    const float* g_lin2w = (const float*)d_in[9];
    const float* g_bias  = (const float*)d_in[10];
    const float* g0_wih  = (const float*)d_in[11];
    const float* g0_whh  = (const float*)d_in[12];
    const float* g0_bih  = (const float*)d_in[13];
    const float* g0_bhh  = (const float*)d_in[14];
    const float* ac_w    = (const float*)d_in[15];
    const float* ac_asrc = (const float*)d_in[16];
    const float* ac_adst = (const float*)d_in[17];
    const float* ac_bias = (const float*)d_in[18];
    const float* g1_wih  = (const float*)d_in[19];
    const float* g1_whh  = (const float*)d_in[20];
    const float* g1_bih  = (const float*)d_in[21];
    const float* g1_bhh  = (const float*)d_in[22];
    const float* mc_w    = (const float*)d_in[23];
    const float* mc_asrc = (const float*)d_in[24];
    const float* mc_adst = (const float*)d_in[25];
    const float* mc_bias = (const float*)d_in[26];
    const float* gm_wih  = (const float*)d_in[27];
    const float* gm_whh  = (const float*)d_in[28];
    const float* gm_bih  = (const float*)d_in[29];
    const float* gm_bhh  = (const float*)d_in[30];
    const float* lin2_w  = (const float*)d_in[31];
    const float* lin2_b  = (const float*)d_in[32];

    const int N = in_sizes[0] / 64;      // 50000 (multiple of 16)
    const int E = in_sizes[1] / 14;      // 800000 (multiple of 16)
    const int G = out_size;              // 512
    const int T = 8;
    const int* srcI = eidx;
    const int* dstI = eidx + E;

    // ---- workspace carve-up
    size_t off = 0;
    auto alloc = [&](size_t bytes) -> void* {
        void* p = (char*)d_ws + off;
        off += (bytes + 255) & ~(size_t)255;
        return p;
    };
    _Float16* Wh_lin1 = (_Float16*)alloc((size_t)96 * 64 * 2);
    _Float16* Wh_g1   = (_Float16*)alloc((size_t)96 * 128 * 2);
    _Float16* Wh_g2   = (_Float16*)alloc((size_t)96 * 96 * 2);
    _Float16* Wh_g0i  = (_Float16*)alloc((size_t)288 * 96 * 2);
    _Float16* Wh_g0h  = (_Float16*)alloc((size_t)288 * 96 * 2);
    _Float16* Wh_ac   = (_Float16*)alloc((size_t)96 * 96 * 2);
    _Float16* Wh_g1i  = (_Float16*)alloc((size_t)288 * 96 * 2);
    _Float16* Wh_g1h  = (_Float16*)alloc((size_t)288 * 96 * 2);
    _Float16* Wh_mc   = (_Float16*)alloc((size_t)96 * 96 * 2);
    _Float16* Wh_gmi  = (_Float16*)alloc((size_t)288 * 96 * 2);
    _Float16* Wh_gmh  = (_Float16*)alloc((size_t)288 * 96 * 2);
    float* x0   = (float*)alloc((size_t)N * 96 * 4);
    float* x1   = (float*)alloc((size_t)N * 96 * 4);
    float* x2   = (float*)alloc((size_t)N * 96 * 4);
    float* hbuf = (float*)alloc((size_t)N * 96 * 4);
    float* xh   = (float*)alloc((size_t)N * 96 * 4);
    float* xm   = (float*)alloc((size_t)N * 96 * 4);
    float* gi   = (float*)alloc((size_t)N * 288 * 4);
    float* gh   = (float*)alloc((size_t)N * 288 * 4);
    float* xr   = (float*)alloc((size_t)N * 4);
    float* ssrc = (float*)alloc((size_t)N * 4);
    float* sdst = (float*)alloc((size_t)N * 4);
    float* asrc = (float*)alloc((size_t)N * 4);
    _Float16* mE = (_Float16*)alloc((size_t)E * 96 * 2);
    float* elogit = (float*)alloc((size_t)E * 4);
    float* ew     = (float*)alloc((size_t)E * 4);
    unsigned* smax = (unsigned*)alloc((size_t)N * 4);
    float* ssum    = (float*)alloc((size_t)N * 4);
    float* out_g = (float*)alloc((size_t)G * 96 * 4);
    float* om    = (float*)alloc((size_t)G * 96 * 4);
    float* h3    = (float*)alloc((size_t)G * 96 * 4);
    float* gi_g  = (float*)alloc((size_t)G * 288 * 4);
    float* gh_g  = (float*)alloc((size_t)G * 288 * 4);
    float* adst  = (float*)alloc((size_t)G * 4);

    auto blk = [](long n, int b) { return (unsigned)((n + b - 1) / b); };
    auto gemm_grid = [&](int M) { return (unsigned)((M / 16 + 3) / 4); }; // 4 waves/block

    // ---- weights -> f16 (g_lin1_w padded 110 -> 128)
    auto cvt = [&](const float* s, _Float16* d, int rows, int scols, int dcols) {
        cvt_w_kernel<<<blk((long)rows * dcols, 256), 256, 0, stream>>>(s, d, rows, scols, dcols);
    };
    cvt(lin1_w, Wh_lin1, 96, 64, 64);
    cvt(g_lin1w, Wh_g1, 96, 110, 128);
    cvt(g_lin2w, Wh_g2, 96, 96, 96);
    cvt(g0_wih, Wh_g0i, 288, 96, 96);  cvt(g0_whh, Wh_g0h, 288, 96, 96);
    cvt(ac_w,   Wh_ac,  96, 96, 96);
    cvt(g1_wih, Wh_g1i, 288, 96, 96);  cvt(g1_whh, Wh_g1h, 288, 96, 96);
    cvt(mc_w,   Wh_mc,  96, 96, 96);
    cvt(gm_wih, Wh_gmi, 288, 96, 96);  cvt(gm_whh, Wh_gmh, 288, 96, 96);

    auto segsoftmax = [&](const float* lg, float* wout, const int* seg, int n, int S) {
        seg_init_kernel<<<blk(S, 256), 256, 0, stream>>>(smax, ssum, S);
        seg_max_kernel <<<blk(n, 256), 256, 0, stream>>>(lg, seg, smax, n);
        seg_exp_kernel <<<blk(n, 256), 256, 0, stream>>>(lg, seg, smax, wout, ssum, n);
        seg_norm_kernel<<<blk(n, 256), 256, 0, stream>>>(wout, seg, ssum, n);
    };
    auto run_gru = [&](const float* xin, const float* hprev, float* outp,
                       const _Float16* Wi, const _Float16* Wh,
                       const float* bih, const float* bhh, int M,
                       float* gibuf, float* ghbuf) {
        gemm_nt_kernel<96, 288, 1, 0><<<gemm_grid(M), 128, 0, stream>>>(xin, Wi, nullptr, gibuf, M);
        gemm_nt_kernel<96, 288, 0, 0><<<gemm_grid(M), 128, 0, stream>>>(hprev, Wh, nullptr, ghbuf, M);
        gru_kernel<<<blk((long)M * 96, 256), 256, 0, stream>>>(gibuf, ghbuf, bih, bhh, hprev, outp, M);
    };

    // ---- 1) x0 = lrelu(x @ lin1_w^T + b)
    gemm_nt_kernel<64, 96, 0, 1><<<gemm_grid(N), 128, 0, stream>>>(x_in, Wh_lin1, lin1_b, x0, N);
    // ---- 2) GATEConv
    rowdot_kernel<<<blk(N, 256), 256, 0, stream>>>(x0, g_att_r, xr, N, 96);
    gate_edge_kernel<<<gemm_grid(E), 128, 0, stream>>>(x0, eattr, srcI, dstI, Wh_g1,
                                                       g_att_l, xr, mE, elogit, E);
    segsoftmax(elogit, ew, dstI, E, N);
    zero_kernel<<<blk((long)N * 96, 256), 256, 0, stream>>>(hbuf, (long)N * 96);
    gate_agg_kernel<<<gemm_grid(E), 128, 0, stream>>>(mE, Wh_g2, ew, dstI, hbuf, E);
    add_bias_kernel<<<blk((long)N * 96, 256), 256, 0, stream>>>(hbuf, g_bias, (long)N * 96);
    run_gru(hbuf, x0, x1, Wh_g0i, Wh_g0h, g0_bih, g0_bhh, N, gi, gh);
    // ---- 3) atom GATConv
    gemm_nt_kernel<96, 96, 0, 0><<<gemm_grid(N), 128, 0, stream>>>(x1, Wh_ac, nullptr, xh, N);
    rowdot_kernel<<<blk(N, 256), 256, 0, stream>>>(xh, ac_asrc, ssrc, N, 96);
    rowdot_kernel<<<blk(N, 256), 256, 0, stream>>>(xh, ac_adst, sdst, N, 96);
    pair_logit_kernel<<<blk(E, 256), 256, 0, stream>>>(ssrc, sdst, srcI, dstI, elogit, E);
    segsoftmax(elogit, ew, dstI, E, N);
    zero_kernel<<<blk((long)N * 96, 256), 256, 0, stream>>>(hbuf, (long)N * 96);
    scatter_rows_kernel<<<blk((long)E * 96, 256), 256, 0, stream>>>(xh, srcI, dstI, ew, hbuf,
                                                                    (long)E * 96);
    add_bias_kernel<<<blk((long)N * 96, 256), 256, 0, stream>>>(hbuf, ac_bias, (long)N * 96);
    run_gru(hbuf, x1, x2, Wh_g1i, Wh_g1h, g1_bih, g1_bhh, N, gi, gh);
    // ---- 4) readout
    zero_kernel<<<blk((long)G * 96, 256), 256, 0, stream>>>(out_g, (long)G * 96);
    scatter_rows_kernel<<<blk((long)N * 96, 256), 256, 0, stream>>>(x2, nullptr, batch, nullptr,
                                                                    out_g, (long)N * 96);
    relu_kernel<<<blk((long)G * 96, 256), 256, 0, stream>>>(out_g, (long)G * 96);
    gemm_nt_kernel<96, 96, 0, 0><<<gemm_grid(N), 128, 0, stream>>>(x2, Wh_mc, nullptr, xm, N);
    rowdot_kernel<<<blk(N, 256), 256, 0, stream>>>(xm, mc_asrc, asrc, N, 96);
    for (int t = 0; t < T; ++t) {
        gemm_nt_kernel<96, 96, 0, 0><<<gemm_grid(G), 128, 0, stream>>>(out_g, Wh_mc, nullptr, om, G);
        rowdot_kernel<<<blk(G, 256), 256, 0, stream>>>(om, mc_adst, adst, G, 96);
        pair_logit_kernel<<<blk(N, 256), 256, 0, stream>>>(asrc, adst, nullptr, batch, elogit, N);
        segsoftmax(elogit, ew, batch, N, G);
        zero_kernel<<<blk((long)G * 96, 256), 256, 0, stream>>>(h3, (long)G * 96);
        scatter_rows_kernel<<<blk((long)N * 96, 256), 256, 0, stream>>>(xm, nullptr, batch, ew, h3,
                                                                        (long)N * 96);
        add_bias_kernel<<<blk((long)G * 96, 256), 256, 0, stream>>>(h3, mc_bias, (long)G * 96);
        run_gru(h3, out_g, out_g, Wh_gmi, Wh_gmh, gm_bih, gm_bhh, G, gi_g, gh_g);
    }
    // ---- 5) final linear -> d_out [G]
    final_dot_kernel<<<blk(G, 256), 256, 0, stream>>>(out_g, lin2_w, lin2_b, (float*)d_out, G);
}
